// FullHistogramAutoencoder_50818053046998
// MI455X (gfx1250) — matrix operand order repaired and checked by reading, not compiled, not run on hardware
//
#include <hip/hip_runtime.h>
#include <hip/hip_bf16.h>

// ---------------------------------------------------------------------------
// CDNA5 (gfx1250): all dense matmuls via v_wmma_f32_16x16x32_f16.
// Weights are pre-staged (once per launch) as transposed f16 [N, Kp] so each
// lane's B fragment is two contiguous b128 loads; A is f32 w/ packed cvt.
// ---------------------------------------------------------------------------

typedef __attribute__((ext_vector_type(16))) _Float16 v16h;
typedef __attribute__((ext_vector_type(8)))  _Float16 v8h;
typedef __attribute__((ext_vector_type(8)))  float    v8f;

#define WAVES_PER_BLOCK 8

// Wt[n*Kp + k] = (k < K) ? (f16)W[k*N + n] : 0
__global__ __launch_bounds__(256) void conv_weight(
    const float* __restrict__ W, _Float16* __restrict__ Wt,
    int K, int N, int Kp)
{
    const int i = blockIdx.x * 256 + threadIdx.x;
    if (i >= N * Kp) return;
    const int n = i / Kp, k = i % Kp;
    Wt[i] = (k < K) ? (_Float16)W[(size_t)k * N + n] : (_Float16)0.f;
}

// Y[M,N] = act(X[M,K] @ W[K,N] + bias[N]) ; act: 0=none, 1=relu
// Wt is the transposed f16 weight [N, Kp]. One 16x(16*NT) C strip per wave.
template <int K, int N, int NT>
__global__ __launch_bounds__(256) void gemm_wmma(
    const float* __restrict__ X, const _Float16* __restrict__ Wt,
    const float* __restrict__ bias, float* __restrict__ Y,
    int M, int act)
{
    constexpr int KP = ((K + 31) / 32) * 32;
    const int lane   = threadIdx.x & 31;
    const int wave   = threadIdx.x >> 5;
    const int tilesN = N / (16 * NT);
    const int tile   = blockIdx.x * WAVES_PER_BLOCK + wave;
    if (tile >= (M >> 4) * tilesN) return;          // wave-uniform exit; EXEC stays all-1
    const int tm     = tile / tilesN;
    const int tn     = tile % tilesN;
    const int laneHi = lane >> 4;                    // 0: lanes 0-15, 1: lanes 16-31
    const int lane15 = lane & 15;

    // Per-lane base pointers; inner-loop addressing is pure immediate offsets.
    const float* __restrict__ Xa = X + (size_t)(tm * 16 + lane15) * K + laneHi * 8;
    const int col0 = tn * (16 * NT) + lane15;
    const _Float16* __restrict__ Wb = Wt + (size_t)col0 * KP + laneHi * 8;

    v8f c[NT];
    #pragma unroll
    for (int t = 0; t < NT; ++t)
        c[t] = (v8f){0.f, 0.f, 0.f, 0.f, 0.f, 0.f, 0.f, 0.f};

    constexpr int KSTEPS = KP / 32;
    for (int ks = 0; ks < KSTEPS; ++ks) {
        v16h a;
        if (K % 32 == 0) {                           // unchecked fast path (folds)
            #pragma unroll
            for (int i = 0; i < 8; ++i) {            // contiguous -> global_load_b128 x2
                a[i]     = (_Float16)Xa[i];
                a[i + 8] = (_Float16)Xa[16 + i];
            }
        } else {                                     // A tail (dec head K=16): folds
            const int k0 = ks * 32;
            #pragma unroll
            for (int i = 0; i < 8; ++i) {
                a[i]     = (k0 + laneHi * 8 + i      < K) ? (_Float16)Xa[i]      : (_Float16)0.f;
                a[i + 8] = (k0 + laneHi * 8 + 16 + i < K) ? (_Float16)Xa[16 + i] : (_Float16)0.f;
            }
        }
        v16h b[NT];
        #pragma unroll
        for (int t = 0; t < NT; ++t) {               // B: contiguous f16, 2x b128 each
            const v8h lo = *(const v8h*)(Wb + t * 16 * KP);      // immediate offsets
            const v8h hi = *(const v8h*)(Wb + t * 16 * KP + 16);
            #pragma unroll
            for (int i = 0; i < 8; ++i) { b[t][i] = lo[i]; b[t][i + 8] = hi[i]; }
        }
        Xa += 32;
        Wb += 32;
        if (ks + 1 < KSTEPS) __builtin_prefetch(Xa, 0, 0);   // global_prefetch_b8
        #pragma unroll
        for (int t = 0; t < NT; ++t)
            c[t] = __builtin_amdgcn_wmma_f32_16x16x32_f16(
                       /*neg_a=*/false, a, /*neg_b=*/false, b[t],
                       /*c_mod=*/(short)0, c[t], /*reuse_a=*/false, /*reuse_b=*/false);
    }

    // C layout: row = r + 8*laneHi, col = lane15 (+ t*16)
    float* __restrict__ Yr = Y + (size_t)(tm * 16 + laneHi * 8) * N + col0;
    #pragma unroll
    for (int t = 0; t < NT; ++t) {
        const float bv = bias ? bias[col0 + t * 16] : 0.f;
        #pragma unroll
        for (int r = 0; r < 8; ++r) {
            float v = c[t][r] + bv;
            if (act == 1) v = fmaxf(v, 0.f);
            Yr[(size_t)r * N + t * 16] = v;          // immediate offsets
        }
    }
}

// Multi-head attention, H=4, hd=128.  One block per (batch*head).
// QKVq / QKVkv: packed [rows,1536] rows = b*TK tokens; Out: [rows,512] head-concat.
template <int TK>
__global__ __launch_bounds__(64) void attn_kernel(
    const float* __restrict__ QKVq, const float* __restrict__ QKVkv,
    float* __restrict__ Out)
{
    extern __shared__ float smem[];
    float* Ks = smem;                 // [TK][128]
    float* Vs = smem + TK * 128;      // [TK][128]
    const int b = blockIdx.x >> 2;
    const int h = blockIdx.x & 3;
    const int t = threadIdx.x;

    for (int idx = threadIdx.x; idx < TK * 128; idx += blockDim.x) {
        const int j = idx >> 7, d = idx & 127;
        const float* row = QKVkv + (size_t)(b * TK + j) * 1536;
        Ks[idx] = row[512  + h * 128 + d];
        Vs[idx] = row[1024 + h * 128 + d];
    }
    __syncthreads();

    if (t < TK) {
        const float* qrow = QKVq + (size_t)(b * TK + t) * 1536 + h * 128;
        float q[128];
        #pragma unroll
        for (int d = 0; d < 128; ++d) q[d] = qrow[d];

        float s[TK];
        float mx = -1e30f;
        #pragma unroll 1
        for (int j = 0; j < TK; ++j) {
            float acc = 0.f;
            #pragma unroll
            for (int d = 0; d < 128; ++d) acc += q[d] * Ks[j * 128 + d];
            acc *= 0.08838834764831845f;             // 1/sqrt(128)
            s[j] = acc;
            mx = fmaxf(mx, acc);
        }
        float sum = 0.f;
        #pragma unroll
        for (int j = 0; j < TK; ++j) { s[j] = __expf(s[j] - mx); sum += s[j]; }
        const float inv = 1.f / sum;

        float* orow = Out + (size_t)(b * TK + t) * 512 + h * 128;
        #pragma unroll 1
        for (int d = 0; d < 128; ++d) {
            float acc = 0.f;
            #pragma unroll
            for (int j = 0; j < TK; ++j) acc += s[j] * Vs[j * 128 + d];
            orow[d] = acc * inv;
        }
    }
}

// out[row,:] = LayerNorm(x[row,:] + r[row,:]) * g + b ; width 512, one wave/row
__global__ __launch_bounds__(256) void ln_residual(
    const float* __restrict__ x, const float* __restrict__ r,
    const float* __restrict__ g, const float* __restrict__ be,
    float* __restrict__ out, int rows)
{
    const int lane = threadIdx.x & 31;
    const int row  = blockIdx.x * 8 + (threadIdx.x >> 5);
    if (row >= rows) return;
    const float* xr = x + (size_t)row * 512;
    const float* rr = r + (size_t)row * 512;

    float v[16];
    float s = 0.f;
    #pragma unroll
    for (int i = 0; i < 16; ++i) { v[i] = xr[lane + i * 32] + rr[lane + i * 32]; s += v[i]; }
    #pragma unroll
    for (int o = 16; o > 0; o >>= 1) s += __shfl_xor(s, o, 32);
    const float mu = s * (1.f / 512.f);

    float var = 0.f;
    #pragma unroll
    for (int i = 0; i < 16; ++i) { const float d = v[i] - mu; var += d * d; }
    #pragma unroll
    for (int o = 16; o > 0; o >>= 1) var += __shfl_xor(var, o, 32);
    const float inv = rsqrtf(var * (1.f / 512.f) + 1e-5f);

    float* orow = out + (size_t)row * 512;
    #pragma unroll
    for (int i = 0; i < 16; ++i) {
        const int cidx = lane + i * 32;
        orow[cidx] = (v[i] - mu) * inv * g[cidx] + be[cidx];
    }
}

// X[b,64,512] = concat(enc[b,32,512], mv[b,32,512], axis=1) + pos[64,512]
__global__ __launch_bounds__(256) void assemble_x(
    const float* __restrict__ enc, const float* __restrict__ mv,
    const float* __restrict__ pos, float* __restrict__ X)
{
    const int i = blockIdx.x * 256 + threadIdx.x;    // over 128*64*512
    const int c = i & 511, t = (i >> 9) & 63, b = i >> 15;
    const float v = (t < 32) ? enc[((size_t)b * 32 + t) * 512 + c]
                             : mv[((size_t)b * 32 + (t - 32)) * 512 + c];
    X[i] = v + pos[t * 512 + c];
}

// latent[i] = X(view [*,8])[i,:] . w[0:8] + b   (128*4096 outputs)
__global__ __launch_bounds__(256) void bottleneck_kernel(
    const float* __restrict__ X, const float* __restrict__ w,
    const float* __restrict__ bs, float* __restrict__ latent)
{
    const int i = blockIdx.x * 256 + threadIdx.x;
    const float* xr = X + (size_t)i * 8;
    float acc = bs[0];
    #pragma unroll
    for (int j = 0; j < 8; ++j) acc += xr[j] * w[j];
    latent[i] = acc;
}

// out[r,c] = t(view [*,16])[r,:] . w[:,c] + b[c]  (c in {0,1})
__global__ __launch_bounds__(256) void dmv_head(
    const float* __restrict__ t, const float* __restrict__ w,
    const float* __restrict__ bb, float* __restrict__ out)
{
    const int i = blockIdx.x * 256 + threadIdx.x;    // over 128*1024*2
    const int c = i & 1, r = i >> 1;
    const float* tr = t + (size_t)r * 16;
    float acc = bb[c];
    #pragma unroll
    for (int j = 0; j < 16; ++j) acc += tr[j] * w[j * 2 + c];
    out[i] = acc;
}

// ---------------------------------------------------------------------------
// Host orchestration
// ---------------------------------------------------------------------------
template <int K, int N>
static inline void launch_gemm(const float* X, const _Float16* Wt, const float* b,
                               float* Y, int M, int act, hipStream_t s)
{
    constexpr int NT = (N % 64 == 0) ? 4 : 1;        // 16x64 strips when possible
    const int tiles  = (M / 16) * (N / (16 * NT));
    const int blocks = (tiles + WAVES_PER_BLOCK - 1) / WAVES_PER_BLOCK;
    gemm_wmma<K, N, NT><<<blocks, 256, 0, s>>>(X, Wt, b, Y, M, act);
}

// stage W[K,N] f32 -> Wt[N,Kp] f16 into bump-allocated workspace
static inline _Float16* stage_weight(const float* W, int K, int N,
                                     _Float16*& bump, hipStream_t s)
{
    const int Kp = ((K + 31) / 32) * 32;
    _Float16* Wt = bump;
    const size_t n = (size_t)N * Kp;
    bump += (n + 127) & ~(size_t)127;                // keep 256B alignment
    conv_weight<<<(int)((n + 255) / 256), 256, 0, s>>>(W, Wt, K, N, Kp);
    return Wt;
}

extern "C" void kernel_launch(void* const* d_in, const int* in_sizes, int n_in,
                              void* d_out, int out_size, void* d_ws, size_t ws_size,
                              hipStream_t stream)
{
    (void)in_sizes; (void)n_in; (void)out_size; (void)ws_size;
    const float* P[128];
    for (int i = 0; i < n_in && i < 128; ++i) P[i] = (const float*)d_in[i];

    const float* hist  = P[0];   // [128,32,32,256] -> [131072,256]
    const float* mv_in = P[1];   // [128,32,64]     -> [4096,64]
    // params, insertion order:
    // 2:enc_w1 3:enc_b1 4:enc_w2 5:enc_b2 6:enc_w3 7:enc_b3 8:ing_w 9:ing_b 10:pos
    // enc layer l: base 11+12l: qkv_w,qkv_b,o_w,o_b,ln1_g,ln1_b,l1_w,l1_b,l2_w,l2_b,ln2_g,ln2_b
    // 35:bot_w 36:bot_b 37:exp_w 38:exp_b
    // dec layer l: base 39+18l: sa_qkv_w,sa_qkv_b,sa_o_w,sa_o_b,ca_qkv_w,ca_qkv_b,ca_o_w,ca_o_b,
    //                           ln1_g,ln1_b,ln2_g,ln2_b,ln3_g,ln3_b,l1_w,l1_b,l2_w,l2_b
    // 75:dmv_w 76:dmv_b 77:dec_w1 78:dec_b1 79:dec_w2 80:dec_b2 81:dec_w3 82:dec_b3

    // ---- workspace layout (floats) ----
    float* ws = (float*)d_ws;
    size_t off = 0;
    float* BUF_A = ws + off; off += (size_t)131072 * 128;  // 16.78M (also FF1 8192x2048)
    float* BUF_B = ws + off; off += (size_t)131072 * 128;  // 16.78M (also dec QKV2)
    float* Xbuf  = ws + off; off += (size_t)8192 * 512;    // encoder tokens [b*64,512]
    float* ENC   = ws + off; off += (size_t)4096 * 512;
    float* MV    = ws + off; off += (size_t)4096 * 512;
    float* QKV   = ws + off; off += (size_t)8192 * 1536;
    float* ATT   = ws + off; off += (size_t)8192 * 512;
    float* ATTO  = ws + off; off += (size_t)8192 * 512;
    float* EXP   = ws + off; off += (size_t)4096 * 512;
    float* DEC   = ws + off; off += (size_t)4096 * 512;
    _Float16* wtBump = (_Float16*)(ws + off);              // staged f16 weights (~30MB)

    // ---- stage all GEMM weights as transposed f16 ----
    const _Float16* Wt_enc1 = stage_weight(P[2], 256, 128, wtBump, stream);
    const _Float16* Wt_enc2 = stage_weight(P[4], 128, 128, wtBump, stream);
    const _Float16* Wt_enc3 = stage_weight(P[6], 128, 16,  wtBump, stream);
    const _Float16* Wt_ing  = stage_weight(P[8], 64,  512, wtBump, stream);
    const _Float16 *Wt_qkv[2], *Wt_o[2], *Wt_l1[2], *Wt_l2[2];
    for (int l = 0; l < 2; ++l) {
        const int e = 11 + l * 12;
        Wt_qkv[l] = stage_weight(P[e + 0], 512, 1536, wtBump, stream);
        Wt_o[l]   = stage_weight(P[e + 2], 512, 512,  wtBump, stream);
        Wt_l1[l]  = stage_weight(P[e + 6], 512, 2048, wtBump, stream);
        Wt_l2[l]  = stage_weight(P[e + 8], 2048, 512, wtBump, stream);
    }
    const _Float16* Wt_exp = stage_weight(P[37], 128, 512, wtBump, stream);
    const _Float16 *Wt_sqkv[2], *Wt_so[2], *Wt_cqkv[2], *Wt_co[2], *Wt_dl1[2], *Wt_dl2[2];
    for (int l = 0; l < 2; ++l) {
        const int d = 39 + l * 18;
        Wt_sqkv[l] = stage_weight(P[d + 0],  512, 1536, wtBump, stream);
        Wt_so[l]   = stage_weight(P[d + 2],  512, 512,  wtBump, stream);
        Wt_cqkv[l] = stage_weight(P[d + 4],  512, 1536, wtBump, stream);
        Wt_co[l]   = stage_weight(P[d + 6],  512, 512,  wtBump, stream);
        Wt_dl1[l]  = stage_weight(P[d + 14], 512, 2048, wtBump, stream);
        Wt_dl2[l]  = stage_weight(P[d + 16], 2048, 512, wtBump, stream);
    }
    const _Float16* Wt_d1 = stage_weight(P[77], 16,  128, wtBump, stream);
    const _Float16* Wt_d2 = stage_weight(P[79], 128, 128, wtBump, stream);
    const _Float16* Wt_d3 = stage_weight(P[81], 128, 256, wtBump, stream);

    // ---- output layout: d[131072,256], mv_dec[131072,2], latent[128*4096] ----
    float* out_d      = (float*)d_out;
    float* out_dmv    = out_d + (size_t)131072 * 256;
    float* out_latent = out_dmv + (size_t)131072 * 2;

    // ===== histogram encoder MLP (131072 tokens) =====
    launch_gemm<256, 128>(hist,  Wt_enc1, P[3], BUF_A, 131072, 1, stream);
    launch_gemm<128, 128>(BUF_A, Wt_enc2, P[5], BUF_B, 131072, 1, stream);
    launch_gemm<128, 16 >(BUF_B, Wt_enc3, P[7], ENC,   131072, 0, stream);  // -> [b,32,512]
    launch_gemm<64, 512 >(mv_in, Wt_ing,  P[9], MV,      4096, 0, stream);
    assemble_x<<<(8192 * 512) / 256, 256, 0, stream>>>(ENC, MV, P[10], Xbuf);

    // ===== encoder layers (8192 tokens, T=64) =====
    for (int l = 0; l < 2; ++l) {
        const int e = 11 + l * 12;
        launch_gemm<512, 1536>(Xbuf, Wt_qkv[l], P[e + 1], QKV, 8192, 0, stream);
        attn_kernel<64><<<128 * 4, 64, 64 * 128 * 2 * sizeof(float), stream>>>(QKV, QKV, ATT);
        launch_gemm<512, 512>(ATT, Wt_o[l], P[e + 3], ATTO, 8192, 0, stream);
        ln_residual<<<8192 / 8, 256, 0, stream>>>(Xbuf, ATTO, P[e + 4], P[e + 5], Xbuf, 8192);
        launch_gemm<512, 2048>(Xbuf,  Wt_l1[l], P[e + 7], BUF_A, 8192, 1, stream);
        launch_gemm<2048, 512>(BUF_A, Wt_l2[l], P[e + 9], ATTO,  8192, 0, stream);
        ln_residual<<<8192 / 8, 256, 0, stream>>>(Xbuf, ATTO, P[e + 10], P[e + 11], Xbuf, 8192);
    }

    // ===== bottleneck -> latent (straight into d_out), expand =====
    bottleneck_kernel<<<(128 * 4096) / 256, 256, 0, stream>>>(Xbuf, P[35], P[36], out_latent);
    launch_gemm<128, 512>(out_latent, Wt_exp, P[38], EXP, 4096, 0, stream); // [b,32,512]
    hipMemcpyAsync(DEC, EXP, (size_t)4096 * 512 * sizeof(float),
                   hipMemcpyDeviceToDevice, stream);                        // y = exp

    // ===== decoder layers (4096 tokens, T=32, cross-attn memory = EXP) =====
    for (int l = 0; l < 2; ++l) {
        const int d = 39 + l * 18;
        // self-attention
        launch_gemm<512, 1536>(DEC, Wt_sqkv[l], P[d + 1], QKV, 4096, 0, stream);
        attn_kernel<32><<<128 * 4, 64, 32 * 128 * 2 * sizeof(float), stream>>>(QKV, QKV, ATT);
        launch_gemm<512, 512>(ATT, Wt_so[l], P[d + 3], ATTO, 4096, 0, stream);
        ln_residual<<<4096 / 8, 256, 0, stream>>>(DEC, ATTO, P[d + 8], P[d + 9], DEC, 4096);
        // cross-attention (Q from y, K/V from memory)
        launch_gemm<512, 1536>(DEC, Wt_cqkv[l], P[d + 5], QKV,   4096, 0, stream);
        launch_gemm<512, 1536>(EXP, Wt_cqkv[l], P[d + 5], BUF_B, 4096, 0, stream);
        attn_kernel<32><<<128 * 4, 64, 32 * 128 * 2 * sizeof(float), stream>>>(QKV, BUF_B, ATT);
        launch_gemm<512, 512>(ATT, Wt_co[l], P[d + 7], ATTO, 4096, 0, stream);
        ln_residual<<<4096 / 8, 256, 0, stream>>>(DEC, ATTO, P[d + 10], P[d + 11], DEC, 4096);
        // feed-forward
        launch_gemm<512, 2048>(DEC,   Wt_dl1[l], P[d + 15], BUF_A, 4096, 1, stream);
        launch_gemm<2048, 512>(BUF_A, Wt_dl2[l], P[d + 17], ATTO,  4096, 0, stream);
        ln_residual<<<4096 / 8, 256, 0, stream>>>(DEC, ATTO, P[d + 12], P[d + 13], DEC, 4096);
    }

    // ===== heads: t = y viewed as [131072,16] =====
    dmv_head<<<(131072 * 2) / 256, 256, 0, stream>>>(DEC, P[75], P[76], out_dmv);
    launch_gemm<16, 128 >(DEC,   Wt_d1, P[78], BUF_A, 131072, 1, stream);
    launch_gemm<128, 128>(BUF_A, Wt_d2, P[80], BUF_B, 131072, 1, stream);
    launch_gemm<128, 256>(BUF_B, Wt_d3, P[82], out_d, 131072, 0, stream);
}
